// LightweightAttn_52235392254346
// MI455X (gfx1250) — matrix operand order repaired and checked by reading, not compiled, not run on hardware
//
#include <hip/hip_runtime.h>

// ---------------------------------------------------------------------------
// Types for CDNA5 WMMA (wave32): v_wmma_f32_16x16x32_bf16
// ---------------------------------------------------------------------------
typedef __attribute__((ext_vector_type(16))) __bf16 bf16x16;
typedef __attribute__((ext_vector_type(8)))  __bf16 bf16x8;
typedef __attribute__((ext_vector_type(8)))  float  v8f;

static __device__ __forceinline__ bf16x16 combine8(bf16x8 lo, bf16x8 hi) {
  return __builtin_shufflevector(lo, hi, 0,1,2,3,4,5,6,7,8,9,10,11,12,13,14,15);
}
static __device__ __forceinline__ v8f zero8() {
  v8f z = {0.f,0.f,0.f,0.f,0.f,0.f,0.f,0.f};
  return z;
}
static __device__ __forceinline__ v8f wmma_bf16(bf16x16 a, bf16x16 b, v8f c) {
  return __builtin_amdgcn_wmma_f32_16x16x32_bf16(false, a, false, b, (short)0, c, false, false);
}

// A-fragment (16x32, bf16): lane half h: elems 0..7 -> K=8h.., elems 8..15 -> K=16+8h..
// src is row-major [rows][32] bf16 in LDS.
static __device__ __forceinline__ bf16x16 frag_a(const __bf16* s, int rowbase, int cn, int ch) {
  const __bf16* p = s + (rowbase + cn) * 32 + 8 * ch;
  bf16x8 lo = *(const bf16x8*)(p);
  bf16x8 hi = *(const bf16x8*)(p + 16);
  return combine8(lo, hi);
}
// B-fragment (32x16, bf16) from B^T layout [n][32]: lane n=cn, elems 0..7 -> K=8*ch.., elems 8..15 -> K=16+8*ch..
static __device__ __forceinline__ bf16x16 frag_bt(const __bf16* s, int colbase, int cn, int ch) {
  const __bf16* p = s + (colbase + cn) * 32 + 8 * ch;
  bf16x8 lo = *(const bf16x8*)(p);
  bf16x8 hi = *(const bf16x8*)(p + 16);
  return combine8(lo, hi);
}

// ---------------------------------------------------------------------------
// Constants
// ---------------------------------------------------------------------------
#define BSZ   4
#define SEQ   2048
#define DM    1024
#define NH    16
#define DH    64
#define MROWS (BSZ * SEQ)     // 8192

// ---------------------------------------------------------------------------
// 1) LayerNorm: one block per row
// ---------------------------------------------------------------------------
__global__ void attn_ln_kernel(const float* __restrict__ x,
                               const float* __restrict__ gamma,
                               const float* __restrict__ beta,
                               float* __restrict__ xn) {
  __shared__ float red[256];
  const int row = blockIdx.x;
  const int tid = threadIdx.x;
  const float* xr = x + (size_t)row * DM;
  float v[4];
  float s = 0.f;
#pragma unroll
  for (int i = 0; i < 4; ++i) { v[i] = xr[tid + i * 256]; s += v[i]; }
  red[tid] = s; __syncthreads();
#pragma unroll
  for (int off = 128; off > 0; off >>= 1) {
    if (tid < off) red[tid] += red[tid + off];
    __syncthreads();
  }
  const float mu = red[0] * (1.0f / DM);
  __syncthreads();
  float s2 = 0.f;
#pragma unroll
  for (int i = 0; i < 4; ++i) { float d = v[i] - mu; s2 += d * d; }
  red[tid] = s2; __syncthreads();
#pragma unroll
  for (int off = 128; off > 0; off >>= 1) {
    if (tid < off) red[tid] += red[tid + off];
    __syncthreads();
  }
  const float rstd = rsqrtf(red[0] * (1.0f / DM) + 1e-5f);
  float* xo = xn + (size_t)row * DM;
#pragma unroll
  for (int i = 0; i < 4; ++i) {
    int c = tid + i * 256;
    xo[c] = (v[i] - mu) * rstd * gamma[c] + beta[c];
  }
}

// ---------------------------------------------------------------------------
// 2) Generic GEMM: out[M,1024] = A[M,1024] @ W[1024,1024] + bias (+ resid)
//    Block tile 128x64, K-step 32, 8 waves (4x2), each wave 32x32 (2x2 WMMA).
// ---------------------------------------------------------------------------
template <bool RESID>
__global__ void attn_gemm_kernel(const float* __restrict__ A,
                                 const float* __restrict__ W,
                                 const float* __restrict__ bias,
                                 const float* __restrict__ resid,
                                 float* __restrict__ out) {
  __shared__ __bf16 sA[128 * 32];   // [row][k]
  __shared__ __bf16 sBt[64 * 32];   // [n][k]
  const int tid  = threadIdx.x;
  const int wave = tid >> 5;
  const int lane = tid & 31;
  const int wm = wave & 3;          // 0..3 row strip
  const int wn = wave >> 2;         // 0..1 col strip
  const int bm = blockIdx.y * 128;
  const int bn = blockIdx.x * 64;
  const int cn = lane & 15, ch = lane >> 4;

  v8f acc[2][2];
#pragma unroll
  for (int i = 0; i < 2; ++i)
#pragma unroll
    for (int j = 0; j < 2; ++j) acc[i][j] = zero8();

  for (int kt = 0; kt < DM / 32; ++kt) {
    const int k0 = kt * 32;
#pragma unroll
    for (int i = 0; i < 16; ++i) {            // 128x32 fp32 -> bf16
      int idx = i * 256 + tid;
      int r = idx >> 5, c = idx & 31;
      sA[r * 32 + c] = (__bf16)A[(size_t)(bm + r) * DM + k0 + c];
    }
#pragma unroll
    for (int i = 0; i < 8; ++i) {             // 32x64 fp32 -> bf16 transposed
      int idx = i * 256 + tid;
      int nn = idx & 63, kk = idx >> 6;
      sBt[nn * 32 + kk] = (__bf16)W[(size_t)(k0 + kk) * DM + bn + nn];
    }
    __syncthreads();
    bf16x16 a0 = frag_a(sA, wm * 32,      cn, ch);
    bf16x16 a1 = frag_a(sA, wm * 32 + 16, cn, ch);
    bf16x16 b0 = frag_bt(sBt, wn * 32,      cn, ch);
    bf16x16 b1 = frag_bt(sBt, wn * 32 + 16, cn, ch);
    acc[0][0] = wmma_bf16(a0, b0, acc[0][0]);
    acc[0][1] = wmma_bf16(a0, b1, acc[0][1]);
    acc[1][0] = wmma_bf16(a1, b0, acc[1][0]);
    acc[1][1] = wmma_bf16(a1, b1, acc[1][1]);
    __syncthreads();
  }
#pragma unroll
  for (int it = 0; it < 2; ++it)
#pragma unroll
    for (int jt = 0; jt < 2; ++jt)
#pragma unroll
      for (int r = 0; r < 8; ++r) {
        int row = bm + wm * 32 + it * 16 + r + 8 * ch;
        int col = bn + wn * 32 + jt * 16 + cn;
        float v = acc[it][jt][r] + bias[col];
        if (RESID) v += resid[(size_t)row * DM + col];
        out[(size_t)row * DM + col] = v;
      }
}

// ---------------------------------------------------------------------------
// 3) RoPE + pack to bf16 head layout [B, H, L, 64]
// ---------------------------------------------------------------------------
__global__ void attn_rope_pack_kernel(const float* __restrict__ Qf,
                                      const float* __restrict__ Kf,
                                      const float* __restrict__ Vf,
                                      __bf16* __restrict__ qh,
                                      __bf16* __restrict__ kh,
                                      __bf16* __restrict__ vh) {
  const int t = blockIdx.x * 256 + threadIdx.x;   // B*L*H*32 threads
  const int i  = t & 31;          // pair index within head
  const int t1 = t >> 5;
  const int hh = t1 & (NH - 1);
  const int t2 = t1 >> 4;
  const int l  = t2 & (SEQ - 1);
  const int b  = t2 >> 11;
  const size_t src = ((size_t)(b * SEQ + l)) * DM + hh * DH + 2 * i;
  const size_t dst = ((size_t)((b * NH + hh) * SEQ + l)) * DH + 2 * i;
  const float inv = __powf(10000.f, -(float)(2 * i) * (1.0f / (float)DH));
  float sn, cs;
  __sincosf((float)l * inv, &sn, &cs);
  float q1 = Qf[src], q2 = Qf[src + 1];
  qh[dst]     = (__bf16)(q1 * cs - q2 * sn);
  qh[dst + 1] = (__bf16)(q1 * sn + q2 * cs);
  float k1 = Kf[src], k2 = Kf[src + 1];
  kh[dst]     = (__bf16)(k1 * cs - k2 * sn);
  kh[dst + 1] = (__bf16)(k1 * sn + k2 * cs);
  vh[dst]     = (__bf16)Vf[src];
  vh[dst + 1] = (__bf16)Vf[src + 1];
}

// ---------------------------------------------------------------------------
// 4) Causal flash attention. 1 wave = 32 query rows; 8 waves/block (no block
//    barriers: waves have divergent causal trip counts, per-wave LDS + in-order
//    DS pipeline + s_wait_dscnt give correctness).
//    __launch_bounds__(256,1): raise the VGPR budget so Q frags / O accum /
//    softmax state stay resident (the previous build spilled to scratch).
// ---------------------------------------------------------------------------
__global__ void __launch_bounds__(256, 1)
attn_flash_kernel(const __bf16* __restrict__ qh,
                  const __bf16* __restrict__ kh,
                  const __bf16* __restrict__ vh,
                  float* __restrict__ attn) {
  __shared__ __bf16 lds_vt[8][DH * 32];  // per-wave V^T tile [dh][key]
  __shared__ __bf16 lds_p[8][32 * 32];   // per-wave P tile   [row][key]
  const int tid  = threadIdx.x;
  const int wave = tid >> 5;
  const int lane = tid & 31;
  const int rb = blockIdx.x & 7;         // 8 row-blocks of 256
  const int bh = blockIdx.x >> 3;
  const int b  = bh >> 4;
  const int hh = bh & 15;
  const int q0 = rb * 256 + wave * 32;
  const int cn = lane & 15, ch = lane >> 4;

  const __bf16* qbase = qh + ((size_t)(b * NH + hh) * SEQ) * DH;
  const __bf16* kbase = kh + ((size_t)(b * NH + hh) * SEQ) * DH;
  const __bf16* vbase = vh + ((size_t)(b * NH + hh) * SEQ) * DH;

  // Q fragments held in registers for the whole pass (A-layout; Dh contiguous)
  bf16x16 aq[2][2];
#pragma unroll
  for (int it = 0; it < 2; ++it) {
    const __bf16* qp = qbase + (size_t)(q0 + it * 16 + cn) * DH;
#pragma unroll
    for (int ks = 0; ks < 2; ++ks) {
      bf16x8 lo = *(const bf16x8*)(qp + ks * 32 + 8 * ch);
      bf16x8 hi = *(const bf16x8*)(qp + ks * 32 + 16 + 8 * ch);
      aq[it][ks] = combine8(lo, hi);
    }
  }

  v8f o[2][4];
  float mst[2][8], lst[2][8];
#pragma unroll
  for (int it = 0; it < 2; ++it) {
#pragma unroll
    for (int c = 0; c < 4; ++c) o[it][c] = zero8();
#pragma unroll
    for (int r = 0; r < 8; ++r) { mst[it][r] = -1e30f; lst[it][r] = 0.f; }
  }

  __bf16* vt = lds_vt[wave];
  __bf16* pp = lds_p[wave];

  const int ntile = q0 / 32 + 1;
  for (int kt = 0; kt < ntile; ++kt) {
    const int key0 = kt * 32;
    // stage V^T (per wave)
#pragma unroll
    for (int i = 0; i < 64; ++i) {
      int idx = i * 32 + lane;
      int dd = idx & 63, key = idx >> 6;
      vt[dd * 32 + key] = vbase[(size_t)(key0 + key) * DH + dd];
    }
    asm volatile("s_wait_dscnt 0x0" ::: "memory");

    // S = Q * K^T  (B-frag straight from global: Dh contiguous per key row)
    v8f s[2][2];
#pragma unroll
    for (int it = 0; it < 2; ++it)
#pragma unroll
      for (int jt = 0; jt < 2; ++jt) s[it][jt] = zero8();
#pragma unroll
    for (int jt = 0; jt < 2; ++jt) {
#pragma unroll
      for (int ks = 0; ks < 2; ++ks) {
        const __bf16* kp = kbase + (size_t)(key0 + jt * 16 + cn) * DH + ks * 32 + 8 * ch;
        bf16x16 bk = combine8(*(const bf16x8*)kp, *(const bf16x8*)(kp + 16));
        s[0][jt] = wmma_bf16(aq[0][ks], bk, s[0][jt]);
        s[1][jt] = wmma_bf16(aq[1][ks], bk, s[1][jt]);
      }
    }

    const float scale = 0.125f;  // 1/sqrt(64)
    const bool diag = (kt == ntile - 1);

    // Fused: mask/scale -> row max -> exp -> P(bf16) to LDS -> l/m -> rescale O.
    // (single sweep so the fp32 S values die right after the exp)
#pragma unroll
    for (int it = 0; it < 2; ++it) {
#pragma unroll
      for (int r = 0; r < 8; ++r) {
        const int row = q0 + it * 16 + r + 8 * ch;
        float v0 = s[it][0][r] * scale;
        float v1 = s[it][1][r] * scale;
        if (diag) {
          if (key0 + cn      > row) v0 = -1e30f;
          if (key0 + 16 + cn > row) v1 = -1e30f;
        }
        float mx = fmaxf(v0, v1);
#pragma unroll
        for (int m = 1; m < 16; m <<= 1) mx = fmaxf(mx, __shfl_xor(mx, m, 32));
        const float mnew = fmaxf(mst[it][r], mx);
        const float corr = __expf(mst[it][r] - mnew);
        const float p0 = __expf(v0 - mnew);
        const float p1 = __expf(v1 - mnew);
        pp[(it * 16 + r + 8 * ch) * 32 + cn]      = (__bf16)p0;
        pp[(it * 16 + r + 8 * ch) * 32 + 16 + cn] = (__bf16)p1;
        float rs = p0 + p1;
#pragma unroll
        for (int m = 1; m < 16; m <<= 1) rs += __shfl_xor(rs, m, 32);
        lst[it][r] = lst[it][r] * corr + rs;
        mst[it][r] = mnew;
#pragma unroll
        for (int c = 0; c < 4; ++c) o[it][c][r] *= corr;
      }
    }
    asm volatile("s_wait_dscnt 0x0" ::: "memory");

    // O += P * V
#pragma unroll
    for (int it = 0; it < 2; ++it) {
      const __bf16* ap = pp + (it * 16 + cn) * 32 + 8 * ch;
      bf16x16 af = combine8(*(const bf16x8*)ap, *(const bf16x8*)(ap + 16));
#pragma unroll
      for (int c = 0; c < 4; ++c) {
        const __bf16* bp = vt + (c * 16 + cn) * 32 + 8 * ch;
        bf16x16 bv = combine8(*(const bf16x8*)bp, *(const bf16x8*)(bp + 16));
        o[it][c] = wmma_bf16(af, bv, o[it][c]);
      }
    }
  }

  // epilogue: normalize, scatter back to [B, L, H*DH] fp32
#pragma unroll
  for (int it = 0; it < 2; ++it)
#pragma unroll
    for (int r = 0; r < 8; ++r) {
      float inv = 1.0f / lst[it][r];
      int row = q0 + it * 16 + r + 8 * ch;
#pragma unroll
      for (int c = 0; c < 4; ++c) {
        int col = hh * DH + c * 16 + cn;
        attn[((size_t)b * SEQ + row) * DM + col] = o[it][c][r] * inv;
      }
    }
}

// ---------------------------------------------------------------------------
// Launch
// ---------------------------------------------------------------------------
extern "C" void kernel_launch(void* const* d_in, const int* in_sizes, int n_in,
                              void* d_out, int out_size, void* d_ws, size_t ws_size,
                              hipStream_t stream) {
  const float* x     = (const float*)d_in[0];
  const float* gamma = (const float*)d_in[1];
  const float* beta  = (const float*)d_in[2];
  const float* Wq    = (const float*)d_in[3];
  const float* bq    = (const float*)d_in[4];
  const float* Wk    = (const float*)d_in[5];
  const float* bk    = (const float*)d_in[6];
  const float* Wv    = (const float*)d_in[7];
  const float* bv    = (const float*)d_in[8];
  const float* Wo    = (const float*)d_in[9];
  const float* bo    = (const float*)d_in[10];
  float* out = (float*)d_out;

  const size_t f32mat  = (size_t)MROWS * DM * sizeof(float);
  const size_t bf16mat = (size_t)MROWS * DM * sizeof(unsigned short);
  char* ws = (char*)d_ws;
  float*  xn = (float*)(ws);
  float*  Qf = (float*)(ws + 1 * f32mat);
  float*  Kf = (float*)(ws + 2 * f32mat);
  float*  Vf = (float*)(ws + 3 * f32mat);
  __bf16* qh = (__bf16*)(ws + 4 * f32mat);
  __bf16* kh = (__bf16*)(ws + 4 * f32mat + 1 * bf16mat);
  __bf16* vh = (__bf16*)(ws + 4 * f32mat + 2 * bf16mat);
  float* attn = Qf;  // Qf dead after rope_pack; reuse for attention output

  attn_ln_kernel<<<MROWS, 256, 0, stream>>>(x, gamma, beta, xn);

  dim3 ggrid(DM / 64, MROWS / 128);
  attn_gemm_kernel<false><<<ggrid, 256, 0, stream>>>(xn, Wq, bq, nullptr, Qf);
  attn_gemm_kernel<false><<<ggrid, 256, 0, stream>>>(xn, Wk, bk, nullptr, Kf);
  attn_gemm_kernel<false><<<ggrid, 256, 0, stream>>>(xn, Wv, bv, nullptr, Vf);

  attn_rope_pack_kernel<<<(BSZ * SEQ * NH * 32) / 256, 256, 0, stream>>>(Qf, Kf, Vf, qh, kh, vh);

  attn_flash_kernel<<<BSZ * NH * (SEQ / 256), 256, 0, stream>>>(qh, kh, vh, attn);

  attn_gemm_kernel<true><<<ggrid, 256, 0, stream>>>(attn, Wo, bo, x, out);
}
